// GroupedQueryAttention_60928406061070
// MI455X (gfx1250) — matrix operand order repaired and checked by reading, not compiled, not run on hardware
//
#include <hip/hip_runtime.h>

// ---------------------------------------------------------------------------
// GroupedQueryAttention for MI455X (gfx1250): bf16 WMMA GEMM pipeline, v3.
// All GEMM operands pre-converted to bf16 (single conversion per element);
// GEMM inner loops are pure data movement + v_wmma_f32_16x16x32_bf16.
// ---------------------------------------------------------------------------

typedef __attribute__((ext_vector_type(16))) __bf16 v16bf;
typedef __attribute__((ext_vector_type(2)))  __bf16 v2bf;
typedef __attribute__((ext_vector_type(2)))  float  v2f;
typedef __attribute__((ext_vector_type(8)))  float  v8f;

#define SQ      2048
#define SK      2048
#define DMODEL  1024
#define NHEADS  16
#define HDIM    64
#define KVDIM   256      // NUM_KV_HEADS * HEAD_DIM
#define NB      2
#define ATT_SCALE 0.125f

// --- f32 -> bf16 (RTNE via fptrunc; lowers to v_cvt_pk_bf16_f32) ------------
__device__ __forceinline__ unsigned int pk2(float x, float y) {
  v2f f; f.x = x; f.y = y;
  v2bf b = __builtin_convertvector(f, v2bf);
  return __builtin_bit_cast(unsigned int, b);
}
__device__ __forceinline__ unsigned short bf1(float x) {
  __bf16 b = (__bf16)x;
  return __builtin_bit_cast(unsigned short, b);
}

// --- WMMA fragment loaders (ISA 7.12.2 layouts), tiles in LDS, ldk = 32 -----
__device__ __forceinline__ v16bf load_a_frag(const unsigned short* sA) {
  const int lane = threadIdx.x & 31;
  const int m = lane & 15;
  const int g = lane >> 4;
  union { v16bf v; uint4 q[2]; } u;
  u.q[0] = *(const uint4*)&sA[m * 32 + g * 8];
  u.q[1] = *(const uint4*)&sA[m * 32 + g * 8 + 16];
  return u.v;
}
__device__ __forceinline__ v16bf load_b_frag(const unsigned short* sB) {
  const int lane = threadIdx.x & 31;
  const int n = lane & 15;
  const int g = lane >> 4;
  union { v16bf v; uint4 q[2]; } u;
  u.q[0] = *(const uint4*)&sB[n * 32 + g * 8];
  u.q[1] = *(const uint4*)&sB[n * 32 + g * 8 + 16];
  return u.v;
}

// ---------------------------------------------------------------------------
// Bulk f32 -> bf16 conversion, 8 elements/thread. grid = n / 2048.
// ---------------------------------------------------------------------------
__global__ __launch_bounds__(256)
void cvt_kernel(const float* __restrict__ src, unsigned short* __restrict__ dst) {
  const size_t i = ((size_t)blockIdx.x * 256 + threadIdx.x) * 8;
  const float4 a = *(const float4*)(src + i);
  const float4 b = *(const float4*)(src + i + 4);
  uint4 o;
  o.x = pk2(a.x, a.y); o.y = pk2(a.z, a.w);
  o.z = pk2(b.x, b.y); o.w = pk2(b.z, b.w);
  *(uint4*)(dst + i) = o;
}

// ---------------------------------------------------------------------------
// C[m,n] = sum_k A[m,k] * B[n,k], A/B bf16 (k-contiguous), C f32 or bf16.
// grid = (M/16, N/64), 128 threads = 4 waves, one 16x16 C tile per wave.
// ---------------------------------------------------------------------------
template <bool OUT_BF16>
__global__ __launch_bounds__(128)
void gemm_bf16_kernel(const unsigned short* __restrict__ A,
                      const unsigned short* __restrict__ B,
                      void* __restrict__ C, int K, int lda, int ldb, int ldc) {
  __shared__ __align__(16) unsigned short sA[16 * 32];
  __shared__ __align__(16) unsigned short sB[64 * 32];
  const int tid  = threadIdx.x;
  const int wv   = tid >> 5;
  const int lane = tid & 31;
  const int m0 = blockIdx.x * 16;
  const int n0 = blockIdx.y * 64;

  const int ar = tid >> 3, ac = (tid & 7) * 4;      // A: uint2 / thread
  const int br = tid >> 2, bc = (tid & 3) * 8;      // B: 2x uint4 / thread
  const unsigned short* Arow  = A + (size_t)(m0 + ar) * lda + ac;
  const unsigned short* Brow0 = B + (size_t)(n0 + br) * ldb + bc;
  const unsigned short* Brow1 = Brow0 + (size_t)32 * ldb;

  union { v8f v; float f[8]; } acc;
#pragma unroll
  for (int r = 0; r < 8; ++r) acc.f[r] = 0.0f;

  for (int k0 = 0; k0 < K; k0 += 32) {
    const uint2 avq = *(const uint2*)(Arow + k0);
    const uint4 bv0 = *(const uint4*)(Brow0 + k0);
    const uint4 bv1 = *(const uint4*)(Brow1 + k0);
    __syncthreads();
    *(uint2*)&sA[ar * 32 + ac]        = avq;
    *(uint4*)&sB[br * 32 + bc]        = bv0;
    *(uint4*)&sB[(32 + br) * 32 + bc] = bv1;
    __syncthreads();
    v16bf a = load_a_frag(sA);
    v16bf b = load_b_frag(sB + wv * 16 * 32);
    acc.v = __builtin_amdgcn_wmma_f32_16x16x32_bf16(false, a, false, b,
                                                    (short)0, acc.v, false, false);
  }

  const int n  = n0 + wv * 16 + (lane & 15);
  const int mh = (lane >> 4) * 8;
#pragma unroll
  for (int r = 0; r < 8; ++r) {
    const size_t o = (size_t)(m0 + r + mh) * ldc + n;
    if (OUT_BF16) ((unsigned short*)C)[o] = bf1(acc.f[r]);
    else          ((float*)C)[o] = acc.f[r];
  }
}

// ---------------------------------------------------------------------------
// scores[b,h,q,s] = (Q_h[q,:] . K_kv[s,:]) * SCALE + mask[q,s]   (Q/K bf16)
// grid = (Sq/16, Sk/64, B*NHEADS); K = HDIM = 64 (2 steps).
// ---------------------------------------------------------------------------
__global__ __launch_bounds__(128)
void scores_kernel(const unsigned short* __restrict__ Qp,
                   const unsigned short* __restrict__ Kp,
                   const float* __restrict__ mask, float* __restrict__ attnW) {
  __shared__ __align__(16) unsigned short sA[16 * 32];
  __shared__ __align__(16) unsigned short sB[64 * 32];
  const int tid  = threadIdx.x;
  const int wv   = tid >> 5;
  const int lane = tid & 31;
  const int z = blockIdx.z;              // b*16 + h
  const int b = z >> 4, h = z & 15;
  const int m0 = blockIdx.x * 16;
  const int n0 = blockIdx.y * 64;

  const int ar = tid >> 3, ac = (tid & 7) * 4;
  const int br = tid >> 2, bc = (tid & 3) * 8;
  const unsigned short* Arow =
      Qp + (size_t)b * SQ * DMODEL + h * HDIM + (size_t)(m0 + ar) * DMODEL + ac;
  const unsigned short* Brow0 =
      Kp + (size_t)b * SK * KVDIM + (h >> 2) * HDIM + (size_t)(n0 + br) * KVDIM + bc;
  const unsigned short* Brow1 = Brow0 + (size_t)32 * KVDIM;
  float* Wout = attnW + (size_t)z * SQ * SK;

  union { v8f v; float f[8]; } acc;
#pragma unroll
  for (int r = 0; r < 8; ++r) acc.f[r] = 0.0f;

#pragma unroll
  for (int k0 = 0; k0 < HDIM; k0 += 32) {
    const uint2 avq = *(const uint2*)(Arow + k0);
    const uint4 bv0 = *(const uint4*)(Brow0 + k0);
    const uint4 bv1 = *(const uint4*)(Brow1 + k0);
    __syncthreads();
    *(uint2*)&sA[ar * 32 + ac]        = avq;
    *(uint4*)&sB[br * 32 + bc]        = bv0;
    *(uint4*)&sB[(32 + br) * 32 + bc] = bv1;
    __syncthreads();
    v16bf a  = load_a_frag(sA);
    v16bf bf = load_b_frag(sB + wv * 16 * 32);
    acc.v = __builtin_amdgcn_wmma_f32_16x16x32_bf16(false, a, false, bf,
                                                    (short)0, acc.v, false, false);
  }

  const int n  = n0 + wv * 16 + (lane & 15);
  const int mh = (lane >> 4) * 8;
#pragma unroll
  for (int r = 0; r < 8; ++r) {
    const int m = m0 + r + mh;
    Wout[(size_t)m * SK + n] = acc.f[r] * ATT_SCALE + mask[(size_t)m * SK + n];
  }
}

// ---------------------------------------------------------------------------
// In-place row softmax over Sk=2048. One 256-thread block per row; float4 I/O.
// ---------------------------------------------------------------------------
__global__ __launch_bounds__(256)
void softmax_kernel(float* __restrict__ W) {
  float* row = W + (size_t)blockIdx.x * SK;
  const int t = threadIdx.x;
  __shared__ float red[256];

  union { float4 q[2]; float f[8]; } v;
  v.q[0] = *(const float4*)(row + t * 8);
  v.q[1] = *(const float4*)(row + t * 8 + 4);

  float mx = -3.4e38f;
#pragma unroll
  for (int i = 0; i < 8; ++i) mx = fmaxf(mx, v.f[i]);
  red[t] = mx; __syncthreads();
  for (int s = 128; s > 0; s >>= 1) {
    if (t < s) red[t] = fmaxf(red[t], red[t + s]);
    __syncthreads();
  }
  mx = red[0]; __syncthreads();

  float sum = 0.0f;
#pragma unroll
  for (int i = 0; i < 8; ++i) { v.f[i] = __expf(v.f[i] - mx); sum += v.f[i]; }
  red[t] = sum; __syncthreads();
  for (int s = 128; s > 0; s >>= 1) {
    if (t < s) red[t] += red[t + s];
    __syncthreads();
  }
  const float inv = 1.0f / red[0];
#pragma unroll
  for (int i = 0; i < 8; ++i) v.f[i] *= inv;
  *(float4*)(row + t * 8)     = v.q[0];
  *(float4*)(row + t * 8 + 4) = v.q[1];
}

// ---------------------------------------------------------------------------
// Transpose V (bf16): Vp (B*Sk, 256) -> VpT (B, 256, Sk). 32x32 LDS tiles.
// grid = (B*Sk/32, 256/32), 256 threads.
// ---------------------------------------------------------------------------
__global__ __launch_bounds__(256)
void transposeV_kernel(const unsigned short* __restrict__ Vp,
                       unsigned short* __restrict__ VpT) {
  __shared__ unsigned short t[32][33];
  const int r0 = blockIdx.x * 32;      // s-row (2048 % 32 == 0, never crosses b)
  const int c0 = blockIdx.y * 32;      // d-col
  const int j  = threadIdx.x & 31;
  const int i4 = (threadIdx.x >> 5) * 4;
#pragma unroll
  for (int e = 0; e < 4; ++e)
    t[i4 + e][j] = Vp[(size_t)(r0 + i4 + e) * KVDIM + c0 + j];
  __syncthreads();
  const int b = r0 >> 11;
  const int s = (r0 & 2047) + j;
#pragma unroll
  for (int e = 0; e < 4; ++e)
    VpT[((size_t)b * KVDIM + c0 + i4 + e) * SK + s] = t[j][i4 + e];
}

// ---------------------------------------------------------------------------
// AO[b,q,h*64+d] (bf16) = sum_s P[b,h,q,s] * VpT[b, kv*64+d, s]
// A = probabilities f32 (each element used exactly once -> convert in-kernel,
// v_cvt_pk_bf16_f32); B = pre-converted bf16.
// grid = (Sq/16, 1, B*NHEADS); N = 64 = 4 waves * 16; K = Sk = 2048.
// ---------------------------------------------------------------------------
__global__ __launch_bounds__(128)
void av_kernel(const float* __restrict__ attnW,
               const unsigned short* __restrict__ VpT,
               unsigned short* __restrict__ AO) {
  __shared__ __align__(16) unsigned short sA[16 * 32];
  __shared__ __align__(16) unsigned short sB[64 * 32];
  const int tid  = threadIdx.x;
  const int wv   = tid >> 5;
  const int lane = tid & 31;
  const int z = blockIdx.z;
  const int b = z >> 4, h = z & 15;
  const int m0 = blockIdx.x * 16;

  const int ar = tid >> 3, ac = (tid & 7) * 4;      // A: float4 / thread
  const int br = tid >> 2, bc = (tid & 3) * 8;      // B: 2x uint4 / thread
  const float* Arow =
      attnW + (size_t)z * SQ * SK + (size_t)(m0 + ar) * SK + ac;
  const unsigned short* Brow0 =
      VpT + ((size_t)b * KVDIM + (h >> 2) * HDIM + br) * SK + bc;
  const unsigned short* Brow1 = Brow0 + (size_t)32 * SK;

  union { v8f v; float f[8]; } acc;
#pragma unroll
  for (int r = 0; r < 8; ++r) acc.f[r] = 0.0f;

  for (int k0 = 0; k0 < SK; k0 += 32) {
    const float4 av = *(const float4*)(Arow + k0);
    const uint4 bv0 = *(const uint4*)(Brow0 + k0);
    const uint4 bv1 = *(const uint4*)(Brow1 + k0);
    __syncthreads();
    { uint2 p; p.x = pk2(av.x, av.y); p.y = pk2(av.z, av.w);
      *(uint2*)&sA[ar * 32 + ac] = p; }
    *(uint4*)&sB[br * 32 + bc]        = bv0;
    *(uint4*)&sB[(32 + br) * 32 + bc] = bv1;
    __syncthreads();
    v16bf a  = load_a_frag(sA);
    v16bf bf = load_b_frag(sB + wv * 16 * 32);
    acc.v = __builtin_amdgcn_wmma_f32_16x16x32_bf16(false, a, false, bf,
                                                    (short)0, acc.v, false, false);
  }

  const int n  = wv * 16 + (lane & 15);      // d within head
  const int mh = (lane >> 4) * 8;
#pragma unroll
  for (int r = 0; r < 8; ++r) {
    const int m = m0 + r + mh;
    AO[(size_t)(b * SQ + m) * DMODEL + h * HDIM + n] = bf1(acc.f[r]);
  }
}

// ---------------------------------------------------------------------------
// Host-side orchestration
// ---------------------------------------------------------------------------
extern "C" void kernel_launch(void* const* d_in, const int* in_sizes, int n_in,
                              void* d_out, int out_size, void* d_ws, size_t ws_size,
                              hipStream_t stream) {
  const float* query = (const float*)d_in[0];   // (B, Sq, D)
  const float* key   = (const float*)d_in[1];   // (B, Sk, D)
  const float* value = (const float*)d_in[2];   // (B, Sk, D)
  const float* mask  = (const float*)d_in[3];   // (Sq, Sk)
  const float* Wq    = (const float*)d_in[4];   // (1024, 1024)
  const float* Wk    = (const float*)d_in[5];   // (256, 1024)
  const float* Wv    = (const float*)d_in[6];   // (256, 1024)
  const float* Wo    = (const float*)d_in[7];   // (1024, 1024)

  float* out   = (float*)d_out;                             // (B, Sq, D)
  float* attnW = (float*)d_out + (size_t)NB * SQ * DMODEL;  // (B, H, Sq, Sk)

  // Workspace (bf16 / ushort elements), ~51 MB total
  unsigned short* Xq  = (unsigned short*)d_ws;               // 4096 x 1024
  unsigned short* Xk  = Xq  + (size_t)4096 * 1024;
  unsigned short* Xv  = Xk  + (size_t)4096 * 1024;
  unsigned short* Wqb = Xv  + (size_t)4096 * 1024;           // 1024 x 1024
  unsigned short* Wkb = Wqb + (size_t)1024 * 1024;           //  256 x 1024
  unsigned short* Wvb = Wkb + (size_t)256 * 1024;
  unsigned short* Wob = Wvb + (size_t)256 * 1024;            // 1024 x 1024
  unsigned short* Qp  = Wob + (size_t)1024 * 1024;           // 4096 x 1024
  unsigned short* Kp  = Qp  + (size_t)4096 * 1024;           // 4096 x 256
  unsigned short* Vp  = Kp  + (size_t)4096 * 256;
  unsigned short* VpT = Vp  + (size_t)4096 * 256;            // (B, 256, 2048)
  unsigned short* AO  = VpT + (size_t)NB * KVDIM * SK;       // 4096 x 1024

  const int Mrows = NB * SQ;  // 4096

  // 0) One-shot f32 -> bf16 conversions (pays conversion VALU exactly once)
  cvt_kernel<<<(4096 * 1024) / 2048, 256, 0, stream>>>(query, Xq);
  cvt_kernel<<<(4096 * 1024) / 2048, 256, 0, stream>>>(key,   Xk);
  cvt_kernel<<<(4096 * 1024) / 2048, 256, 0, stream>>>(value, Xv);
  cvt_kernel<<<(1024 * 1024) / 2048, 256, 0, stream>>>(Wq, Wqb);
  cvt_kernel<<<(256  * 1024) / 2048, 256, 0, stream>>>(Wk, Wkb);
  cvt_kernel<<<(256  * 1024) / 2048, 256, 0, stream>>>(Wv, Wvb);
  cvt_kernel<<<(1024 * 1024) / 2048, 256, 0, stream>>>(Wo, Wob);

  // 1) Projections: X @ W^T  (bf16 in, bf16 out)
  gemm_bf16_kernel<true><<<dim3(Mrows / 16, DMODEL / 64), 128, 0, stream>>>(
      Xq, Wqb, Qp, DMODEL, DMODEL, DMODEL, DMODEL);
  gemm_bf16_kernel<true><<<dim3(Mrows / 16, KVDIM / 64), 128, 0, stream>>>(
      Xk, Wkb, Kp, DMODEL, DMODEL, DMODEL, KVDIM);
  gemm_bf16_kernel<true><<<dim3(Mrows / 16, KVDIM / 64), 128, 0, stream>>>(
      Xv, Wvb, Vp, DMODEL, DMODEL, DMODEL, KVDIM);

  // 1b) Transpose V so the P*V GEMM is k-contiguous on both operands
  transposeV_kernel<<<dim3(Mrows / 32, KVDIM / 32), 256, 0, stream>>>(Vp, VpT);

  // 2) Scores = scale * Q K^T + mask -> attn_weights buffer (raw scores)
  scores_kernel<<<dim3(SQ / 16, SK / 64, NB * NHEADS), 128, 0, stream>>>(
      Qp, Kp, mask, attnW);

  // 3) Row softmax in place
  softmax_kernel<<<dim3(NB * NHEADS * SQ), 256, 0, stream>>>(attnW);

  // 4) O = P V  -> AO (bf16)
  av_kernel<<<dim3(SQ / 16, 1, NB * NHEADS), 128, 0, stream>>>(attnW, VpT, AO);

  // 5) out = O @ Wo^T  (f32 out)
  gemm_bf16_kernel<false><<<dim3(Mrows / 16, DMODEL / 64), 128, 0, stream>>>(
      AO, Wob, out, DMODEL, DMODEL, DMODEL, DMODEL);
}
